// Attention_1494648619518
// MI455X (gfx1250) — 8-bit matrix operand order repaired and checked by reading, compiled, not run on hardware
//
#include <hip/hip_runtime.h>
#include <hip/hip_bf16.h>

typedef int v8i __attribute__((ext_vector_type(8)));

#define HEADS  12
#define DH     64
#define SEQ    2048
#define BATCH  4
#define DIM    768
#define NQKV   2304
#define MROWS  (BATCH * SEQ)
#define QMAXF  15.0f

__device__ __forceinline__ int ld_i32(const signed char* p) {
  return *(const int*)p;
}

__device__ __forceinline__ v8i load_a_frag(const signed char* base, int stride, int lane) {
  const int m  = lane & 15;
  const int hi = (lane >> 4) << 3;
  v8i a;
#pragma unroll
  for (int v = 0; v < 8; ++v) {
    const int k = ((v >> 1) << 4) + hi + ((v & 1) << 2);
    a[v] = ld_i32(base + m * stride + k);
  }
  return a;
}

__device__ __forceinline__ v8i load_b_frag(const signed char* base, int stride, int lane) {
  const int n  = lane & 15;
  const int hi = (lane >> 4) << 3;
  v8i b;
#pragma unroll
  for (int v = 0; v < 8; ++v) {
    const int k = ((v >> 1) << 4) + hi + ((v & 1) << 2);
    b[v] = ld_i32(base + n * stride + k);
  }
  return b;
}

__device__ __forceinline__ float quant_code(float x, float inv_s, float zr) {
  return fminf(QMAXF, fmaxf(0.0f, rintf(x * inv_s) + zr)) - zr;
}

__device__ __forceinline__ void async_b128(signed char* lds_dst,
                                           const signed char* sbase, int voff) {
  const unsigned lds_off = (unsigned)(uintptr_t)lds_dst;
  asm volatile("global_load_async_to_lds_b128 %0, %1, %2"
               :: "v"(lds_off), "v"(voff), "s"(sbase)
               : "memory");
}
__device__ __forceinline__ void wait_async_le10() {
  asm volatile("s_wait_asynccnt 0xa" ::: "memory");
}
__device__ __forceinline__ void wait_async_0() {
  asm volatile("s_wait_asynccnt 0x0" ::: "memory");
}

__device__ __forceinline__ void stageA(signed char* dst, const signed char* gbase,
                                       int goff, int lane) {
  const int seg = lane & 3;
  const int row = lane >> 2;
#pragma unroll
  for (int i = 0; i < 2; ++i) {
    const int rr = i * 8 + row;
    async_b128(dst + rr * 64 + seg * 16, gbase, goff + rr * DIM + seg * 16);
  }
}
__device__ __forceinline__ void stageB(signed char* dst, const signed char* gbase,
                                       int goff, int lane) {
  const int seg = lane & 3;
  const int row = lane >> 2;
#pragma unroll
  for (int j = 0; j < 8; ++j) {
    const int nn = j * 8 + row;
    async_b128(dst + nn * 64 + seg * 16, gbase, goff + nn * DIM + seg * 16);
  }
}

__device__ __forceinline__ void gemm_core(const signed char* __restrict__ A, int row0,
                                          const signed char* __restrict__ W, int col0,
                                          signed char (*sA)[16 * 64],
                                          signed char (*sB)[64 * 64],
                                          int lane, v8i acc[4]) {
  stageA(sA[0], A, row0 * DIM, lane);
  stageB(sB[0], W, col0 * DIM, lane);
#pragma unroll 1
  for (int c = 0; c < DIM / 64; ++c) {
    const int cur = c & 1;
    if (c + 1 < DIM / 64) {
      stageA(sA[cur ^ 1], A, row0 * DIM + (c + 1) * 64, lane);
      stageB(sB[cur ^ 1], W, col0 * DIM + (c + 1) * 64, lane);
      wait_async_le10();
    } else {
      wait_async_0();
    }
    const v8i a = load_a_frag(sA[cur], 64, lane);
#pragma unroll
    for (int s = 0; s < 4; ++s) {
      const v8i b = load_b_frag(sB[cur] + s * 16 * 64, 64, lane);
      acc[s] = __builtin_amdgcn_wmma_i32_16x16x64_iu8(true, a, true, b, acc[s],
                                                      false, false);
    }
  }
}

__global__ __launch_bounds__(256)
void quantize_kernel(const float* __restrict__ in, signed char* __restrict__ out,
                     int n, const float* __restrict__ s, const float* __restrict__ z) {
  const int i = blockIdx.x * blockDim.x + threadIdx.x;
  if (i >= n) return;
  const float inv_s = 1.0f / s[0];
  const float zr    = rintf(z[0]);
  out[i] = (signed char)quant_code(in[i], inv_s, zr);
}

__global__ __launch_bounds__(32)
void qkv_gemm_kernel(const signed char* __restrict__ xc,
                     const signed char* __restrict__ w1c,
                     signed char* __restrict__ qc,
                     signed char* __restrict__ kc,
                     signed char* __restrict__ vct,
                     const float* __restrict__ sx1, const float* __restrict__ sw1,
                     const float* __restrict__ sq,  const float* __restrict__ zq,
                     const float* __restrict__ sk,  const float* __restrict__ zk,
                     const float* __restrict__ sv,  const float* __restrict__ zv) {
  __shared__ signed char sA[2][16 * 64];
  __shared__ signed char sB[2][64 * 64];

  const int lane = threadIdx.x;
  const int mt   = blockIdx.y;
  const int ng   = blockIdx.x;
  const int row0 = mt * 16;
  const int col0 = ng * 64;

  v8i acc[4] = {};
  gemm_core(xc, row0, w1c, col0, sA, sB, lane, acc);

  const float sab   = sx1[0] * sw1[0];
  const int   which = col0 / 768;
  const int   h     = (col0 % 768) / DH;
  float inv_s, zr;
  if (which == 0)      { inv_s = 1.0f / sq[0]; zr = rintf(zq[0]); }
  else if (which == 1) { inv_s = 1.0f / sk[0]; zr = rintf(zk[0]); }
  else                 { inv_s = 1.0f / sv[0]; zr = rintf(zv[0]); }

  const int rhi = (lane >> 4) << 3;
#pragma unroll
  for (int s = 0; s < 4; ++s) {
    const int d = s * 16 + (lane & 15);
#pragma unroll
    for (int r = 0; r < 8; ++r) {
      const int   grow = row0 + r + rhi;
      const int   b    = grow / SEQ;
      const int   n    = grow % SEQ;
      const int   bh   = b * HEADS + h;
      const float val  = sab * (float)acc[s][r];
      const signed char code = (signed char)quant_code(val, inv_s, zr);
      if (which == 0)      qc[((size_t)bh * SEQ + n) * DH + d] = code;
      else if (which == 1) kc[((size_t)bh * SEQ + n) * DH + d] = code;
      else                 vct[((size_t)bh * DH + d) * SEQ + n] = code;
    }
  }
}

__global__ __launch_bounds__(32)
void attn_kernel(const signed char* __restrict__ qc,
                 const signed char* __restrict__ kc,
                 const signed char* __restrict__ vct,
                 signed char* __restrict__ xc2,
                 const float* __restrict__ sq, const float* __restrict__ sk,
                 const float* __restrict__ sa, const float* __restrict__ za,
                 const float* __restrict__ sv,
                 const float* __restrict__ sx2, const float* __restrict__ zx2) {
  __shared__ short sdots[16 * SEQ];

  const int lane = threadIdx.x;
  const int tile = blockIdx.x;
  const int bh   = tile / (SEQ / 16);
  const int mt   = tile % (SEQ / 16);
  const int row0 = mt * 16;
  const int rhi  = (lane >> 4) << 3;

  const v8i aq = load_a_frag(qc + ((size_t)bh * SEQ + row0) * DH, DH, lane);
  int rmax[8];
#pragma unroll
  for (int r = 0; r < 8; ++r) rmax[r] = -2147483647;

#pragma unroll 1
  for (int t = 0; t < SEQ / 16; ++t) {
    const v8i bk = load_b_frag(kc + ((size_t)bh * SEQ + t * 16) * DH, DH, lane);
    v8i d = {};
    d = __builtin_amdgcn_wmma_i32_16x16x64_iu8(true, aq, true, bk, d, false, false);
    const int col = t * 16 + (lane & 15);
#pragma unroll
    for (int r = 0; r < 8; ++r) {
      sdots[(r + rhi) * SEQ + col] = (short)d[r];
      rmax[r] = max(rmax[r], d[r]);
    }
  }
#pragma unroll
  for (int r = 0; r < 8; ++r) {
#pragma unroll
    for (int m = 1; m < 16; m <<= 1)
      rmax[r] = max(rmax[r], __shfl_xor(rmax[r], m));
  }
  __syncthreads();

  const int myrow = lane & 15;
  int own = rmax[0];
#pragma unroll
  for (int r = 1; r < 8; ++r) own = ((myrow & 7) == r) ? rmax[r] : own;
  const int oth = __shfl_xor(own, 16);
  const int mx  = ((myrow >> 3) == (lane >> 4)) ? own : oth;

  const int cbeg = (lane >> 4) * (SEQ / 2);
  const float cscale = sq[0] * sk[0] * 0.125f;
  float fsum = 0.0f;
  const short4* prow = (const short4*)&sdots[myrow * SEQ + cbeg];
#pragma unroll 4
  for (int c = 0; c < SEQ / 2 / 4; ++c) {
    const short4 v4 = prow[c];
    fsum += __expf(cscale * (float)((int)v4.x - mx));
    fsum += __expf(cscale * (float)((int)v4.y - mx));
    fsum += __expf(cscale * (float)((int)v4.z - mx));
    fsum += __expf(cscale * (float)((int)v4.w - mx));
  }
  fsum += __shfl_xor(fsum, 16);

  const float zar    = rintf(za[0]);
  const float ascale = 1.0f / (sa[0] * fsum);

  v8i oacc[4] = {};
#pragma unroll 1
  for (int ch = 0; ch < SEQ / 64; ++ch) {
    v8i af;
#pragma unroll
    for (int v = 0; v < 8; ++v) {
      const int mb = ch * 64 + ((v >> 1) << 4) + rhi + ((v & 1) << 2);
      const short4 v4 = *(const short4*)&sdots[myrow * SEQ + mb];
      unsigned packed = 0;
      const int dd[4] = {v4.x, v4.y, v4.z, v4.w};
#pragma unroll
      for (int bb = 0; bb < 4; ++bb) {
        const float e = __expf(cscale * (float)(dd[bb] - mx));
        const float c = fminf(QMAXF, fmaxf(0.0f, rintf(e * ascale) + zar)) - zar;
        packed |= ((unsigned)(unsigned char)(signed char)c) << (8 * bb);
      }
      af[v] = (int)packed;
    }
#pragma unroll
    for (int s = 0; s < 4; ++s) {
      const v8i bv = load_b_frag(vct + ((size_t)bh * DH + s * 16) * SEQ + ch * 64,
                                 SEQ, lane);
      oacc[s] = __builtin_amdgcn_wmma_i32_16x16x64_iu8(true, af, true, bv, oacc[s],
                                                       false, false);
    }
  }

  const float osc    = sa[0] * sv[0];
  const float inv_s2 = 1.0f / sx2[0];
  const float z2r    = rintf(zx2[0]);
  const int   b      = bh / HEADS;
  const int   h      = bh % HEADS;
#pragma unroll
  for (int s = 0; s < 4; ++s) {
    const int d = s * 16 + (lane & 15);
#pragma unroll
    for (int r = 0; r < 8; ++r) {
      const int   n   = row0 + r + rhi;
      const float val = osc * (float)oacc[s][r];
      xc2[((size_t)(b * SEQ + n)) * DIM + h * DH + d] =
          (signed char)quant_code(val, inv_s2, z2r);
    }
  }
}

__global__ __launch_bounds__(32)
void out_gemm_kernel(const signed char* __restrict__ xc2,
                     const signed char* __restrict__ w2c,
                     const float* __restrict__ bb2,
                     float* __restrict__ out,
                     const float* __restrict__ sx2, const float* __restrict__ sw2) {
  __shared__ signed char sA[2][16 * 64];
  __shared__ signed char sB[2][64 * 64];

  const int lane = threadIdx.x;
  const int mt   = blockIdx.y;
  const int ng   = blockIdx.x;
  const int row0 = mt * 16;
  const int col0 = ng * 64;

  v8i acc[4] = {};
  gemm_core(xc2, row0, w2c, col0, sA, sB, lane, acc);

  const float sab = sx2[0] * sw2[0];
  const int   rhi = (lane >> 4) << 3;
#pragma unroll
  for (int s = 0; s < 4; ++s) {
    const int   col  = col0 + s * 16 + (lane & 15);
    const float bias = bb2[col];
#pragma unroll
    for (int r = 0; r < 8; ++r)
      out[(size_t)(row0 + r + rhi) * DIM + col] = sab * (float)acc[s][r] + bias;
  }
}

extern "C" void kernel_launch(void* const* d_in, const int* in_sizes, int n_in,
                              void* d_out, int out_size, void* d_ws, size_t ws_size,
                              hipStream_t stream) {
  const float* x   = (const float*)d_in[0];
  const float* ww1 = (const float*)d_in[1];
  const float* ww2 = (const float*)d_in[2];
  const float* bb2 = (const float*)d_in[3];
  const float* sx1 = (const float*)d_in[4];
  const float* zx1 = (const float*)d_in[5];
  const float* sw1 = (const float*)d_in[6];
  const float* zw1 = (const float*)d_in[7];
  const float* sq  = (const float*)d_in[8];
  const float* zq  = (const float*)d_in[9];
  const float* sk  = (const float*)d_in[10];
  const float* zk  = (const float*)d_in[11];
  const float* sa  = (const float*)d_in[12];
  const float* za  = (const float*)d_in[13];
  const float* sv  = (const float*)d_in[14];
  const float* zv  = (const float*)d_in[15];
  const float* sx2 = (const float*)d_in[16];
  const float* zx2 = (const float*)d_in[17];
  const float* sw2 = (const float*)d_in[18];
  const float* zw2 = (const float*)d_in[19];
  float* out = (float*)d_out;

  char* ws = (char*)d_ws;
  size_t off = 0;
  auto alloc = [&](size_t bytes) {
    char* p = ws + off;
    off += (bytes + 255) & ~(size_t)255;
    return p;
  };
  signed char* xc  = (signed char*)alloc((size_t)MROWS * DIM);
  signed char* w1c = (signed char*)alloc((size_t)NQKV * DIM);
  signed char* w2c = (signed char*)alloc((size_t)DIM * DIM);
  signed char* qc  = (signed char*)alloc((size_t)BATCH * HEADS * SEQ * DH);
  signed char* kc  = (signed char*)alloc((size_t)BATCH * HEADS * SEQ * DH);
  signed char* vct = (signed char*)alloc((size_t)BATCH * HEADS * DH * SEQ);
  signed char* xc2 = (signed char*)alloc((size_t)MROWS * DIM);
  (void)ws_size;

  {
    int n = MROWS * DIM;
    quantize_kernel<<<(n + 255) / 256, 256, 0, stream>>>(x, xc, n, sx1, zx1);
  }
  {
    int n = NQKV * DIM;
    quantize_kernel<<<(n + 255) / 256, 256, 0, stream>>>(ww1, w1c, n, sw1, zw1);
  }
  {
    int n = DIM * DIM;
    quantize_kernel<<<(n + 255) / 256, 256, 0, stream>>>(ww2, w2c, n, sw2, zw2);
  }

  qkv_gemm_kernel<<<dim3(NQKV / 64, MROWS / 16), 32, 0, stream>>>(
      xc, w1c, qc, kc, vct, sx1, sw1, sq, zq, sk, zk, sv, zv);

  attn_kernel<<<BATCH * HEADS * (SEQ / 16), 32, 0, stream>>>(
      qc, kc, vct, xc2, sq, sk, sa, za, sv, sx2, zx2);

  out_gemm_kernel<<<dim3(DIM / 64, MROWS / 16), 32, 0, stream>>>(
      xc2, w2c, bb2, out, sx2, sw2);
}
